// RouteGeneratorModel_2010044694550
// MI455X (gfx1250) — matrix operand order repaired and checked by reading, not compiled, not run on hardware
//
#include <hip/hip_runtime.h>
#include <math.h>

// ---------------------------------------------------------------------------
// RouteGeneratorModel for MI455X (gfx1250).
//   comp    = sigmoid(relu(x@W1c+b1c)@W2c+b2c)          [2048,1]
//   budget  = clip(floor(512 + 7680*mean(comp)^2), 512, 8192)
//   (one-shot) Xb/W1b/W2b = bf16(x / Ws1 / Ws2)
//   H       = relu(Xb@W1b+bs1)      bf16 WMMA + TDM     [2048,1024]
//   scores  = H@W2b + bs2 + 0.1*noise  (WMMA + TDM)     [2048,32768]
//   per row: sort 32768 (key|idx u64) descending in 256KB LDS (320KB/WGP),
//            masked softmax over first `budget`, emit indices + weights.
//
// GEMM structure: 256 threads = 8 waves, block tile 64(M) x 128(N), K-step 32.
//   Wave tile 16x64 -> 4 x v_wmma_f32_16x16x32_bf16 per K-step.
//   A/B tiles staged to LDS by the Tensor Data Mover (wave 0 issues
//   tensor_load_to_lds with a hand-built D#), double buffered with
//   s_wait_tensorcnt + barrier.
//
// d_out (4B elems): [indices 2048*8192 int][budget int][weights 2048*8192 f32][comp 2048 f32]
// d_ws: comp | budget | H(4MB) | Xb(8MB) | W1b(4MB) | W2b(64MB) | scores(256MB) ~ 336MB
// ---------------------------------------------------------------------------

typedef __attribute__((ext_vector_type(16))) __bf16 v16bf;
typedef __attribute__((ext_vector_type(4)))  __bf16 v4bf;
typedef __attribute__((ext_vector_type(8)))  float  v8f;
typedef unsigned int u32x4 __attribute__((ext_vector_type(4)));
typedef int          i32x8 __attribute__((ext_vector_type(8)));
typedef int          i32x4 __attribute__((ext_vector_type(4)));

#define INPUT_DIM 2048
#define POOL      32768
#define HID       1024
#define BATCH     2048
#define MAXP      8192
#define MINP      512
#define NOISE_STD 0.1f

// ---------------------------------------------------------------------------
// Tensor Data Mover: 2-D bf16 tile (tile_d0 contiguous elems x tile_d1 rows)
// from global (row pitch stride0 elems) into LDS at lds_off. D# per ISA 8.3/8.4.
// This toolchain exposes the 6-arg builtin:
//   (u32x4 group0, i32x8 group1, i32x4 group2, i32x4 group3, i32x8 extra, i32 cpol)
// ---------------------------------------------------------------------------
#if __has_builtin(__builtin_amdgcn_tensor_load_to_lds)
#define HAVE_TDM 1
__device__ __forceinline__ void tdm_load_2d_bf16(unsigned lds_off, const void* gaddr,
                                                 unsigned tensor_d0, unsigned tensor_d1,
                                                 unsigned tile_d0, unsigned tile_d1,
                                                 unsigned long long stride0) {
  unsigned long long ga = (unsigned long long)(size_t)gaddr;
  u32x4 g0;
  g0[0] = 1u;                                                   // count=1 (valid)
  g0[1] = lds_off;                                              // lds_addr
  g0[2] = (unsigned)(ga & 0xffffffffu);                         // global_addr[31:0]
  g0[3] = (unsigned)((ga >> 32) & 0x01ffffffu) | (2u << 30);    // ga[56:32] | type=2
  i32x8 g1;
  g1[0] = (1 << 16);                                            // data_size=1 (2B)
  g1[1] = (int)((tensor_d0 & 0xffffu) << 16);                   // tensor_dim0[15:0]
  g1[2] = (int)(((tensor_d0 >> 16) & 0xffffu) |
                ((tensor_d1 & 0xffffu) << 16));                 // td0[31:16] | td1[15:0]
  g1[3] = (int)(((tensor_d1 >> 16) & 0xffffu) |
                ((tile_d0 & 0xffffu) << 16));                   // td1[31:16] | tile_d0
  g1[4] = (int)(tile_d1 & 0xffffu);                             // tile_d1 | tile_d2=0
  g1[5] = (int)(stride0 & 0xffffffffu);                         // dim0_stride[31:0]
  g1[6] = (int)((stride0 >> 32) & 0xffffu);                     // dim0_stride[47:32]
  g1[7] = 0;
  i32x4 z4 = {0, 0, 0, 0};
  i32x8 z8 = {0, 0, 0, 0, 0, 0, 0, 0};
  __builtin_amdgcn_tensor_load_to_lds(g0, g1, z4, z4, z8, 0);
}
#else
#define HAVE_TDM 0
#endif

__device__ __forceinline__ unsigned lds_offset_of(const void* p) {
  return (unsigned)(size_t)p;   // low 32 bits of the LDS-aperture flat address
}

// order-preserving float<->uint transforms
__device__ __forceinline__ unsigned order_key(float f) {
  unsigned u = __float_as_uint(f);
  return (u & 0x80000000u) ? ~u : (u | 0x80000000u);
}
__device__ __forceinline__ float key_score(unsigned t) {
  unsigned u = (t & 0x80000000u) ? (t & 0x7fffffffu) : ~t;
  return __uint_as_float(u);
}

// ---------------- one-shot f32 -> bf16 conversion ---------------------------
__global__ void cvt_bf16_kernel(const float* __restrict__ src,
                                __bf16* __restrict__ dst, int n4) {
  int i = blockIdx.x * blockDim.x + threadIdx.x;
  if (i < n4) {
    float4 f = ((const float4*)src)[i];
    v4bf o;
    o[0] = (__bf16)f.x; o[1] = (__bf16)f.y; o[2] = (__bf16)f.z; o[3] = (__bf16)f.w;
    ((v4bf*)dst)[i] = o;
  }
}

// ---------------- shared GEMM mainloop (bf16 x bf16 -> f32 accum) -----------
union Frag16 { uint4 q[2]; v16bf v; };

template <int KDIM, int NDIM>
__device__ __forceinline__ void gemm_bf16_mainloop(
    const __bf16* __restrict__ Ag, const __bf16* __restrict__ Bg, int bm, int bn,
    unsigned short (&As)[2][64 * 32], unsigned short (&Bs)[2][32 * 128], v8f (&acc)[4]) {
  const int tid  = threadIdx.x;
  const int lane = tid & 31;
  const int wave = tid >> 5;
  const int m    = lane & 15;
  const int half = lane >> 4;
  const int tmw  = (wave & 3) * 16;   // wave row offset in block tile
  const int tnw  = (wave >> 2) * 64;  // wave col offset in block tile

#if HAVE_TDM
  const bool loader = (tid < 32);     // wave 0 drives the TDM
  unsigned aOff0 = lds_offset_of(&As[0][0]), aOff1 = lds_offset_of(&As[1][0]);
  unsigned bOff0 = lds_offset_of(&Bs[0][0]), bOff1 = lds_offset_of(&Bs[1][0]);
  if (loader) {
    tdm_load_2d_bf16(aOff0, Ag + (size_t)bm * KDIM, KDIM, BATCH, 32, 64, KDIM);
    tdm_load_2d_bf16(bOff0, Bg + bn, NDIM, KDIM, 128, 32, NDIM);
    __builtin_amdgcn_s_wait_tensorcnt(0);
  }
  __syncthreads();
#else
  {
    int t8 = tid * 8;
    *(uint4*)&As[0][t8] = *(const uint4*)(Ag + (size_t)(bm + (t8 >> 5)) * KDIM + (t8 & 31));
    int t16 = tid * 16;
    *(uint4*)&Bs[0][t16]     = *(const uint4*)(Bg + (size_t)(t16 >> 7) * NDIM + bn + (t16 & 127));
    *(uint4*)&Bs[0][t16 + 8] = *(const uint4*)(Bg + (size_t)(t16 >> 7) * NDIM + bn + (t16 & 127) + 8);
  }
  __syncthreads();
#endif

  int buf = 0;
  for (int k0 = 0; k0 < KDIM; k0 += 32) {
    const int nk = k0 + 32;
#if HAVE_TDM
    if (loader && nk < KDIM) {   // prefetch next K-step into the other buffer
      tdm_load_2d_bf16(buf ? aOff0 : aOff1, Ag + (size_t)bm * KDIM + nk, KDIM, BATCH, 32, 64, KDIM);
      tdm_load_2d_bf16(buf ? bOff0 : bOff1, Bg + (size_t)nk * NDIM + bn, NDIM, KDIM, 128, 32, NDIM);
    }
#endif
    // A fragment: lane m -> row tmw+m; two contiguous 16B chunks (ISA 7.12.2)
    const char* ar = (const char*)&As[buf][(tmw + m) * 32] + half * 16;
    Frag16 fa;
    fa.q[0] = *(const uint4*)ar;
    fa.q[1] = *(const uint4*)(ar + 32);
#pragma unroll
    for (int nt = 0; nt < 4; ++nt) {
      // B fragment: lane column n = m; k = 8*half+s (s<8), 16+8*half+(s-8) (s>=8)
      const unsigned short* bc = &Bs[buf][tnw + nt * 16 + m];
      Frag16 fb;
      unsigned short* fs = (unsigned short*)&fb;
#pragma unroll
      for (int s = 0; s < 16; ++s) fs[s] = bc[((s & 8) * 2 + 8 * half + (s & 7)) * 128];
      acc[nt] = __builtin_amdgcn_wmma_f32_16x16x32_bf16(false, fa.v, false, fb.v,
                                                        (short)0, acc[nt], false, false);
    }
#if HAVE_TDM
    if (loader) __builtin_amdgcn_s_wait_tensorcnt(0);
    __syncthreads();
    buf ^= 1;
#else
    __syncthreads();
    if (nk < KDIM) {
      int t8 = tid * 8;
      *(uint4*)&As[0][t8] = *(const uint4*)(Ag + (size_t)(bm + (t8 >> 5)) * KDIM + nk + (t8 & 31));
      int t16 = tid * 16;
      *(uint4*)&Bs[0][t16]     = *(const uint4*)(Bg + (size_t)(nk + (t16 >> 7)) * NDIM + bn + (t16 & 127));
      *(uint4*)&Bs[0][t16 + 8] = *(const uint4*)(Bg + (size_t)(nk + (t16 >> 7)) * NDIM + bn + (t16 & 127) + 8);
    }
    __syncthreads();
#endif
  }
}

// ---------------- GEMM1: H = relu(Xb @ W1b + bs1) ---------------------------
__global__ void __launch_bounds__(256)
gemm1_kernel(const __bf16* __restrict__ Xb, const __bf16* __restrict__ W1b,
             const float* __restrict__ bias, __bf16* __restrict__ H) {
  __shared__ __align__(16) unsigned short As[2][64 * 32];
  __shared__ __align__(16) unsigned short Bs[2][32 * 128];
  const int nb = HID / 128;
  const int bm = (blockIdx.x / nb) * 64;
  const int bn = (blockIdx.x % nb) * 128;
  v8f acc[4] = {};
  gemm_bf16_mainloop<INPUT_DIM, HID>(Xb, W1b, bm, bn, As, Bs, acc);
  const int lane = threadIdx.x & 31, wave = threadIdx.x >> 5;
  const int m = lane & 15, half = lane >> 4;
  const int tmw = (wave & 3) * 16, tnw = (wave >> 2) * 64;
#pragma unroll
  for (int nt = 0; nt < 4; ++nt) {
    const int n = bn + tnw + nt * 16 + m;
    const float bn_ = bias[n];
#pragma unroll
    for (int r = 0; r < 8; ++r) {
      const int mm = bm + tmw + r + 8 * half;
      H[(size_t)mm * HID + n] = (__bf16)fmaxf(acc[nt][r] + bn_, 0.f);
    }
  }
}

// ---------------- GEMM2: scores = H @ W2b + bs2 + 0.1*noise -----------------
__global__ void __launch_bounds__(256)
gemm2_kernel(const __bf16* __restrict__ H, const __bf16* __restrict__ W2b,
             const float* __restrict__ bias, const float* __restrict__ noise,
             float* __restrict__ scores) {
  __shared__ __align__(16) unsigned short As[2][64 * 32];
  __shared__ __align__(16) unsigned short Bs[2][32 * 128];
  const int nb = POOL / 128;
  const int bm = (blockIdx.x / nb) * 64;
  const int bn = (blockIdx.x % nb) * 128;
  v8f acc[4] = {};
  gemm_bf16_mainloop<HID, POOL>(H, W2b, bm, bn, As, Bs, acc);
  const int lane = threadIdx.x & 31, wave = threadIdx.x >> 5;
  const int m = lane & 15, half = lane >> 4;
  const int tmw = (wave & 3) * 16, tnw = (wave >> 2) * 64;
#pragma unroll
  for (int nt = 0; nt < 4; ++nt) {
    const int n = bn + tnw + nt * 16 + m;
    const float bn_ = bias[n];
#pragma unroll
    for (int r = 0; r < 8; ++r) {
      const int mm = bm + tmw + r + 8 * half;
      const size_t o = (size_t)mm * POOL + n;
      scores[o] = acc[nt][r] + bn_ + NOISE_STD * noise[o];
    }
  }
}

// ---------------- complexity net: one row per block, 128 threads ------------
__global__ void comp_kernel(const float* __restrict__ x, const float* __restrict__ W1c,
                            const float* __restrict__ b1c, const float* __restrict__ W2c,
                            const float* __restrict__ b2c, float* __restrict__ comp,
                            float* __restrict__ out_comp) {
  __shared__ float xrow[INPUT_DIM];
  __shared__ float red[128];
  const int row = blockIdx.x;
  const int j = threadIdx.x;
  for (int i = j; i < INPUT_DIM; i += 128) xrow[i] = x[(size_t)row * INPUT_DIM + i];
  __syncthreads();
  float a = 0.f;
  for (int i = 0; i < INPUT_DIM; ++i) a = fmaf(xrow[i], W1c[(size_t)i * 128 + j], a);
  red[j] = fmaxf(a + b1c[j], 0.f) * W2c[j];
  __syncthreads();
  for (int s = 64; s > 0; s >>= 1) { if (j < s) red[j] += red[j + s]; __syncthreads(); }
  if (j == 0) {
    float c = 1.f / (1.f + expf(-(red[0] + b2c[0])));
    comp[row] = c;
    out_comp[row] = c;
  }
}

// ---------------- budget scalar --------------------------------------------
__global__ void budget_kernel(const float* __restrict__ comp, int* __restrict__ budget_ws,
                              int* __restrict__ out_budget) {
  __shared__ float red[1024];
  int t = threadIdx.x;
  red[t] = comp[t] + comp[t + 1024];
  __syncthreads();
  for (int s = 512; s > 0; s >>= 1) { if (t < s) red[t] += red[t + s]; __syncthreads(); }
  if (t == 0) {
    float mean = red[0] / (float)BATCH;
    int b = (int)floorf((float)MINP + (float)(MAXP - MINP) * mean * mean);
    b = min(max(b, MINP), MAXP);
    *budget_ws = b;
    *out_budget = b;
  }
}

// ------- per-row: full bitonic sort (32768 u64) in 256KB LDS + softmax ------
__global__ void __launch_bounds__(1024)
topk_softmax_kernel(const float* __restrict__ scores, const int* __restrict__ budget_ptr,
                    int* __restrict__ out_indices, float* __restrict__ out_weights) {
  extern __shared__ unsigned long long kv[];  // POOL entries (key<<32 | idx)
  __shared__ float red[1024];
  const int row = blockIdx.x, tid = threadIdx.x, nthr = 1024;
  const float* srow = scores + (size_t)row * POOL;
  for (int i = tid; i < POOL; i += nthr)
    kv[i] = ((unsigned long long)order_key(srow[i]) << 32) | (unsigned)i;
  __syncthreads();
  for (unsigned k = 2; k <= (unsigned)POOL; k <<= 1) {
    for (unsigned j = k >> 1; j > 0; j >>= 1) {
      for (unsigned i = tid; i < (unsigned)POOL; i += nthr) {
        unsigned ixj = i ^ j;
        if (ixj > i) {
          unsigned long long a = kv[i], b = kv[ixj];
          bool desc = ((i & k) == 0);
          if (desc ? (a < b) : (a > b)) { kv[i] = b; kv[ixj] = a; }
        }
      }
      __syncthreads();
    }
  }
  const int budget = *budget_ptr;
  const float maxv = key_score((unsigned)(kv[0] >> 32));
  float local = 0.f;
  for (int i = tid; i < budget; i += nthr)
    local += expf(key_score((unsigned)(kv[i] >> 32)) - maxv);
  red[tid] = local;
  __syncthreads();
  for (int s = 512; s > 0; s >>= 1) { if (tid < s) red[tid] += red[tid + s]; __syncthreads(); }
  const float inv_sum = 1.f / red[0];
  for (int i = tid; i < MAXP; i += nthr) {
    unsigned long long e = kv[i];
    out_indices[(size_t)row * MAXP + i] = (int)(e & 0xffffffffu);
    float w = 0.f;
    if (i < budget) w = expf(key_score((unsigned)(e >> 32)) - maxv) * inv_sum;
    out_weights[(size_t)row * MAXP + i] = w;
  }
}

// ---------------------------------------------------------------------------
extern "C" void kernel_launch(void* const* d_in, const int* in_sizes, int n_in,
                              void* d_out, int out_size, void* d_ws, size_t ws_size,
                              hipStream_t stream) {
  (void)in_sizes; (void)n_in; (void)out_size; (void)ws_size;
  const float* x     = (const float*)d_in[0];
  const float* noise = (const float*)d_in[1];
  const float* W1c   = (const float*)d_in[2];
  const float* b1c   = (const float*)d_in[3];
  const float* W2c   = (const float*)d_in[4];
  const float* b2c   = (const float*)d_in[5];
  const float* Ws1   = (const float*)d_in[6];
  const float* bs1   = (const float*)d_in[7];
  const float* Ws2   = (const float*)d_in[8];
  const float* bs2   = (const float*)d_in[9];

  const size_t Ni = (size_t)BATCH * MAXP;
  int*   out_indices = (int*)d_out;
  int*   out_budget  = (int*)d_out + Ni;
  float* out_weights = (float*)d_out + Ni + 1;
  float* out_comp    = out_weights + Ni;

  char* ws = (char*)d_ws;
  float*  ws_comp   = (float*)ws;                               // 2048 f32
  int*    ws_budget = (int*)(ws + 8192);
  size_t  off = 8448;
  __bf16* ws_H      = (__bf16*)(ws + off); off += (size_t)BATCH * HID * 2;
  __bf16* ws_Xb     = (__bf16*)(ws + off); off += (size_t)BATCH * INPUT_DIM * 2;
  __bf16* ws_W1b    = (__bf16*)(ws + off); off += (size_t)INPUT_DIM * HID * 2;
  __bf16* ws_W2b    = (__bf16*)(ws + off); off += (size_t)HID * POOL * 2;
  float*  ws_scores = (float*)(ws + off);                       // 2048*32768 f32

  // one-shot bf16 conversions (vectorized, 4 elems/thread)
  cvt_bf16_kernel<<<(BATCH * INPUT_DIM / 4 + 255) / 256, 256, 0, stream>>>(x, ws_Xb,
                                                                 BATCH * INPUT_DIM / 4);
  cvt_bf16_kernel<<<(INPUT_DIM * HID / 4 + 255) / 256, 256, 0, stream>>>(Ws1, ws_W1b,
                                                                 INPUT_DIM * HID / 4);
  cvt_bf16_kernel<<<(HID * POOL / 4 + 255) / 256, 256, 0, stream>>>(Ws2, ws_W2b,
                                                                 HID * POOL / 4);

  comp_kernel<<<BATCH, 128, 0, stream>>>(x, W1c, b1c, W2c, b2c, ws_comp, out_comp);
  budget_kernel<<<1, 1024, 0, stream>>>(ws_comp, ws_budget, out_budget);

  gemm1_kernel<<<(BATCH / 64) * (HID / 128), 256, 0, stream>>>(ws_Xb, ws_W1b, bs1, ws_H);
  gemm2_kernel<<<(BATCH / 64) * (POOL / 128), 256, 0, stream>>>(ws_H, ws_W2b, bs2, noise,
                                                                ws_scores);

  // 256 KB dynamic LDS per block (CDNA5: up to 320 KB/workgroup)
  (void)hipFuncSetAttribute(reinterpret_cast<const void*>(&topk_softmax_kernel),
                            hipFuncAttributeMaxDynamicSharedMemorySize, POOL * 8);
  topk_softmax_kernel<<<BATCH, 1024, POOL * 8, stream>>>(ws_scores, ws_budget,
                                                         out_indices, out_weights);
}
